// GuideModule_16277926052487
// MI455X (gfx1250) — matrix operand order repaired and checked by reading, not compiled, not run on hardware
//
#include <hip/hip_runtime.h>
#include <hip/hip_bf16.h>

typedef __attribute__((ext_vector_type(16))) _Float16 v16h;
typedef __attribute__((ext_vector_type(8)))  float    v8f;
typedef _Float16 half_t;

// Problem constants (from setup_inputs)
constexpr int B_   = 4;
constexpr int C_   = 256;
constexpr int T_   = 16;
constexpr int H_   = 14;
constexpr int W_   = 14;
constexpr int THW_ = T_ * H_ * W_;        // 3136
constexpr int MTOT = B_ * THW_;           // 12544
constexpr int KTAPS = 27;
constexpr int KSTEPS = KTAPS * (C_ / 32); // 216 K-chunks of 32
constexpr int MT   = 64;                  // M tile per workgroup
constexpr int NT_D = 16;                  // 256 / 16 N-tiles (deform)
constexpr int NT_O = 6;                   // 96 / 16 N-tiles (offset, 81 padded)

// ---------------------------------------------------------------------------
// Repack deform_w (C,C,27) f32 -> f16 B-fragments in per-lane, contiguous-k WMMA fragment
// layout: frag[((s*NT + nt)*32 + lane)*16 + vj*2 + hh], K index
// kappa = 32*s + (lane<16?0:16) + 2*vj + hh ; kappa = kk*256 + c ; n = nt*16+lane%16
// ---------------------------------------------------------------------------
__global__ __launch_bounds__(256) void repack_deform(const float* __restrict__ dw,
                                                     half_t* __restrict__ bf) {
    int e = blockIdx.x * 256 + threadIdx.x;
    constexpr int total = KSTEPS * NT_D * 512;
    if (e >= total) return;
    int hh   = e & 1;
    int vj   = (e >> 1) & 7;
    int lane = (e >> 4) & 31;
    int nt   = (e >> 9) & 15;
    int s    = e >> 13;
    int kap  = 32 * s + (lane & 16) + 2 * vj + hh;
    int n    = nt * 16 + (lane & 15);
    int kk   = kap >> 8;
    int c    = kap & 255;
    bf[e] = (half_t)dw[(n * C_ + c) * KTAPS + kk];
}

__global__ __launch_bounds__(256) void repack_off(const float* __restrict__ ow,
                                                  half_t* __restrict__ bf) {
    int e = blockIdx.x * 256 + threadIdx.x;
    constexpr int total = KSTEPS * NT_O * 512;
    if (e >= total) return;
    int w512 = e & 511;
    int snt  = e >> 9;
    int s    = snt / NT_O;
    int nt   = snt % NT_O;
    int lane = w512 >> 4;
    int vj   = (w512 >> 1) & 7;
    int hh   = w512 & 1;
    int kap  = 32 * s + (lane & 16) + 2 * vj + hh;
    int n    = nt * 16 + (lane & 15);
    int kk   = kap >> 8;
    int c    = kap & 255;
    bf[e] = (n < 81) ? (half_t)ow[(n * C_ + c) * KTAPS + kk] : (half_t)0.0f;
}

// ---------------------------------------------------------------------------
// Mask: sigmoid(1x1x1 conv over C) — one wave32 per voxel, shuffle reduce.
// ---------------------------------------------------------------------------
__global__ __launch_bounds__(256) void mask_kernel(const float* __restrict__ x,
                                                   const float* __restrict__ mw,
                                                   const float* __restrict__ mb,
                                                   float* __restrict__ mask_out) {
    int wid = threadIdx.x >> 5, lane = threadIdx.x & 31;
    int g = blockIdx.x * 8 + wid;
    int b = g / THW_, r = g % THW_;
    float s = 0.f;
    for (int c = lane; c < C_; c += 32) s += x[(size_t)(b * C_ + c) * THW_ + r] * mw[c];
    for (int off = 16; off; off >>= 1) s += __shfl_xor(s, off);
    if (lane == 0) mask_out[g] = 1.f / (1.f + __expf(-(s + mb[0])));
}

// ---------------------------------------------------------------------------
// WMMA A-fragment gather from LDS tile As[64][32] (contiguous k: element s holds k = 16*(lane/16) + s)
// ---------------------------------------------------------------------------
__device__ inline v16h load_a_frag(const half_t As[MT][32], int am, int kb) {
    v16h a;
#pragma unroll
    for (int j = 0; j < 8; ++j) {
        int ko = 2 * kb + 2 * j;
        a[2 * j]     = As[am][ko];
        a[2 * j + 1] = As[am][ko + 1];
    }
    return a;
}

// ---------------------------------------------------------------------------
// Offset conv GEMM: M=12544, K=6912, N=96(81). 64-voxel tiles, 8 waves,
// im2col A in LDS, f16 WMMA, f32 accumulate, bias epilogue -> offs (ws).
// ---------------------------------------------------------------------------
__global__ __launch_bounds__(256) void offset_gemm(const float* __restrict__ x,
                                                   const half_t* __restrict__ bf,
                                                   const float* __restrict__ ob,
                                                   float* __restrict__ offs) {
    __shared__ half_t As[MT][32];
    __shared__ short vb[MT], vt[MT], vh[MT], vw[MT];
    int tid = threadIdx.x;
    int m0 = blockIdx.x * MT;
    if (tid < MT) {
        int g = m0 + tid;
        int b = g / THW_, r = g % THW_;
        vb[tid] = (short)b;
        vt[tid] = (short)(r / (H_ * W_));
        vh[tid] = (short)((r / W_) % H_);
        vw[tid] = (short)(r % W_);
    }
    __syncthreads();
    int wid = tid >> 5, lane = tid & 31;
    int mstrip = wid & 3;
    int ntb = (wid >> 2) * 3;
    v8f acc[3] = {};
    int am = mstrip * 16 + (lane & 15);
    int kb = (lane & 16) >> 1;    // 0 or 8

    for (int s = 0; s < KSTEPS; ++s) {
        int kk = s >> 3;
        int c0 = (s & 7) << 5;
        int dt = kk / 9 - 1, dh = (kk / 3) % 3 - 1, dwv = kk % 3 - 1;
        __syncthreads();
#pragma unroll
        for (int i = 0; i < 8; ++i) {
            int idx = tid + i * 256;
            int m = idx >> 5, cc = idx & 31;
            int ti = vt[m] + dt, hi = vh[m] + dh, wi = vw[m] + dwv;
            float v = 0.f;
            if ((unsigned)ti < (unsigned)T_ && (unsigned)hi < (unsigned)H_ &&
                (unsigned)wi < (unsigned)W_)
                v = x[(size_t)(vb[m] * C_ + c0 + cc) * THW_ + (ti * H_ + hi) * W_ + wi];
            As[m][cc] = (half_t)v;
        }
        __syncthreads();
        v16h a = load_a_frag(As, am, kb);
#pragma unroll
        for (int i = 0; i < 3; ++i) {
            int nt = ntb + i;
            v16h bfr = *((const v16h*)(bf + (((size_t)(s * NT_O + nt) * 32 + lane) << 4)));
            acc[i] = __builtin_amdgcn_wmma_f32_16x16x32_f16(false, a, false, bfr,
                                                            (short)0, acc[i], false, false);
        }
    }
    int mrow = mstrip * 16 + ((lane & 16) >> 1); // +8 for lanes 16..31
#pragma unroll
    for (int i = 0; i < 3; ++i) {
        int n = (ntb + i) * 16 + (lane & 15);
        if (n < 81) {
            float bias = ob[n];
#pragma unroll
            for (int j = 0; j < 8; ++j)
                offs[(size_t)(m0 + mrow + j) * 81 + n] = acc[i][j] + bias;
        }
    }
}

// ---------------------------------------------------------------------------
// Deform GEMM: M=12544, K=6912, N=256. Trilinear-gathered A built on the fly
// from precomputed per-(voxel,tap) corner weights (f16) + u16 indices in LDS.
// Fused epilogue: +bias, BN, mask*warp + x residual, ReLU.
// ---------------------------------------------------------------------------
__global__ __launch_bounds__(256) void deform_gemm(
    const float* __restrict__ x, const half_t* __restrict__ bf,
    const float* __restrict__ offs, const float* __restrict__ db,
    const float* __restrict__ gamma, const float* __restrict__ beta,
    const float* __restrict__ mean, const float* __restrict__ var,
    const float* __restrict__ mask, float* __restrict__ out) {
    __shared__ half_t As[MT][32];
    __shared__ half_t cwgt[MT][KTAPS][8];
    __shared__ unsigned short cidx[MT][KTAPS][8];
    __shared__ short vb[MT], vt[MT], vh[MT], vw[MT];
    __shared__ int vr[MT];
    int tid = threadIdx.x;
    int m0 = blockIdx.x * MT;
    if (tid < MT) {
        int g = m0 + tid;
        int b = g / THW_, r = g % THW_;
        vb[tid] = (short)b; vr[tid] = r;
        vt[tid] = (short)(r / (H_ * W_));
        vh[tid] = (short)((r / W_) % H_);
        vw[tid] = (short)(r % W_);
    }
    __syncthreads();
    // Precompute 8 trilinear corners per (voxel, tap)
    for (int task = tid; task < MT * KTAPS; task += 256) {
        int m = task / KTAPS, kk = task % KTAPS;
        const float* op = offs + (size_t)(m0 + m) * 81 + kk * 3;
        float ct = (float)vt[m] + (float)(kk / 9) - 1.f + op[0];
        float ch = (float)vh[m] + (float)((kk / 3) % 3) - 1.f + op[1];
        float cw = (float)vw[m] + (float)(kk % 3) - 1.f + op[2];
        float t0f = floorf(ct), h0f = floorf(ch), w0f = floorf(cw);
        float ft = ct - t0f, fh = ch - h0f, fw = cw - w0f;
        int t0 = (int)t0f, h0 = (int)h0f, w0 = (int)w0f;
#pragma unroll
        for (int j = 0; j < 8; ++j) {
            int it = (j >> 2) & 1, ih = (j >> 1) & 1, iw = j & 1;
            int ti = t0 + it, hi = h0 + ih, wi = w0 + iw;
            float wgt = (it ? ft : 1.f - ft) * (ih ? fh : 1.f - fh) * (iw ? fw : 1.f - fw);
            bool valid = ((unsigned)ti < (unsigned)T_) && ((unsigned)hi < (unsigned)H_) &&
                         ((unsigned)wi < (unsigned)W_);
            int tc = min(max(ti, 0), T_ - 1), hc = min(max(hi, 0), H_ - 1),
                wc = min(max(wi, 0), W_ - 1);
            cwgt[m][kk][j] = (half_t)(valid ? wgt : 0.f);
            cidx[m][kk][j] = (unsigned short)((tc * H_ + hc) * W_ + wc);
        }
    }
    __syncthreads();
    int wid = tid >> 5, lane = tid & 31;
    int mstrip = wid & 3, ntb = (wid >> 2) * 8;
    v8f acc[8] = {};
    int am = mstrip * 16 + (lane & 15);
    int kb = (lane & 16) >> 1;

    for (int s = 0; s < KSTEPS; ++s) {
        int kk = s >> 3, c0 = (s & 7) << 5;
        __syncthreads();
#pragma unroll
        for (int i = 0; i < 8; ++i) {
            int idx = tid + i * 256;
            int m = idx >> 5, cc = idx & 31;
            const float* xb = x + (size_t)(vb[m] * C_ + c0 + cc) * THW_;
            float v = 0.f;
#pragma unroll
            for (int j = 0; j < 8; ++j)
                v += (float)cwgt[m][kk][j] * xb[cidx[m][kk][j]];
            As[m][cc] = (half_t)v;
        }
        __syncthreads();
        v16h a = load_a_frag(As, am, kb);
#pragma unroll
        for (int i = 0; i < 8; ++i) {
            int nt = ntb + i;
            v16h bfr = *((const v16h*)(bf + (((size_t)(s * NT_D + nt) * 32 + lane) << 4)));
            acc[i] = __builtin_amdgcn_wmma_f32_16x16x32_f16(false, a, false, bfr,
                                                            (short)0, acc[i], false, false);
        }
    }
    int mrow = mstrip * 16 + ((lane & 16) >> 1);
#pragma unroll
    for (int i = 0; i < 8; ++i) {
        int n = (ntb + i) * 16 + (lane & 15);
        float inv = gamma[n] * rsqrtf(var[n] + 1e-5f);
        float bias = db[n], mu = mean[n], bet = beta[n];
#pragma unroll
        for (int j = 0; j < 8; ++j) {
            int m = mrow + j;
            int g = m0 + m;
            size_t oidx = (size_t)(vb[m] * C_ + n) * THW_ + vr[m];
            float warp = acc[i][j] + bias;
            warp = (warp - mu) * inv + bet;
            out[oidx] = fmaxf(mask[g] * warp + x[oidx], 0.f);
        }
    }
}

// ---------------------------------------------------------------------------
extern "C" void kernel_launch(void* const* d_in, const int* in_sizes, int n_in,
                              void* d_out, int out_size, void* d_ws, size_t ws_size,
                              hipStream_t stream) {
    (void)in_sizes; (void)n_in; (void)out_size; (void)ws_size;
    const float* x     = (const float*)d_in[0];
    const float* ow    = (const float*)d_in[1];
    const float* ob    = (const float*)d_in[2];
    const float* dw    = (const float*)d_in[3];
    const float* db    = (const float*)d_in[4];
    const float* gamma = (const float*)d_in[5];
    const float* beta  = (const float*)d_in[6];
    const float* mean  = (const float*)d_in[7];
    const float* var   = (const float*)d_in[8];
    const float* mw    = (const float*)d_in[9];
    const float* mb    = (const float*)d_in[10];

    float* out      = (float*)d_out;
    float* mask_out = out + (size_t)B_ * C_ * THW_;   // second tuple output

    // Workspace layout (all 256B-aligned offsets): offs | Bfrag_deform | Bfrag_off
    char* ws = (char*)d_ws;
    float*  offs = (float*)ws;                                    // 12544*81*4  = 4,064,256 B
    half_t* bfd  = (half_t*)(ws + 4064256);                       // 216*16*512*2 = 3,538,944 B
    half_t* bfo  = (half_t*)(ws + 4064256 + 3538944);             // 216*6*512*2  = 1,327,104 B

    constexpr int totD = KSTEPS * NT_D * 512;
    constexpr int totO = KSTEPS * NT_O * 512;
    repack_deform<<<(totD + 255) / 256, 256, 0, stream>>>(dw, bfd);
    repack_off  <<<(totO + 255) / 256, 256, 0, stream>>>(ow, bfo);
    mask_kernel <<<MTOT / 8, 256, 0, stream>>>(x, mw, mb, mask_out);
    offset_gemm <<<MTOT / MT, 256, 0, stream>>>(x, bfo, ob, offs);
    deform_gemm <<<MTOT / MT, 256, 0, stream>>>(x, bfd, offs, db, gamma, beta, mean, var,
                                                mask_out, out);
}
